// SelfAttnPropagation_58067957842067
// MI455X (gfx1250) — matrix operand order repaired and checked by reading, not compiled, not run on hardware
//
#include <hip/hip_runtime.h>

typedef __attribute__((ext_vector_type(16))) _Float16 v16h;
typedef __attribute__((ext_vector_type(8)))  _Float16 v8h;
typedef __attribute__((ext_vector_type(8)))  float    v8f;
typedef __attribute__((ext_vector_type(4)))  float    v4f;

#define BB   2
#define CC   128
#define NN   (96*96)       // 9216 tokens
#define CVV  4
#define SCALE 0.08838834764831845f   // 1/sqrt(128)

#define GROUP   64         // keys staged per LDS tile
#define NGROUPS (NN/GROUP) // 144
#define NWAVES  4
#define ROWS_PER_BLK (16*NWAVES)     // 64
#define NBLK_PER_B (NN/ROWS_PER_BLK) // 144

// ---------------- workspace layout (bytes) ----------------
#define WQ16_OFF 0u
#define WK16_OFF 32768u
#define Q16_OFF  65536u                       // [B][N][C] f16 = 4718592
#define K16_OFF  (65536u + 4718592u)          // 4784128
#define VT_OFF   (4784128u + 4718592u)        // 9502720, Vt [B][N][CV] f32 (294912)

// Load A-operand (16x32 f16, this lane's row) from a row-major f16 row pointer.
// ISA A layout: lane half h, element j -> K = q*32 + (j/8)*16 + h*8 + (j%8)
__device__ __forceinline__ void load_a_rowmajor(const _Float16* row, int half, v16h a[4]) {
#pragma unroll
  for (int q = 0; q < 4; ++q) {
    v8h lo = *(const v8h*)(row + q * 32 + half * 8);
    v8h hi = *(const v8h*)(row + q * 32 + 16 + half * 8);
    a[q] = __builtin_shufflevector(lo, hi, 0,1,2,3,4,5,6,7,8,9,10,11,12,13,14,15);
  }
}

// ---------------- kernel 0: weights f32 -> f16 ----------------
__global__ void k_cvt_weights(const float* __restrict__ Wq, const float* __restrict__ Wk,
                              _Float16* __restrict__ wq16, _Float16* __restrict__ wk16) {
  int i = blockIdx.x * 256 + threadIdx.x;
  if (i < CC * CC) {
    wq16[i] = (_Float16)Wq[i];
    wk16[i] = (_Float16)Wk[i];
  }
}

// ---------------- kernel 0b: V [b][cv][n] -> Vt [b][n][cv] ----------------
__global__ void k_vt(const float* __restrict__ val, float* __restrict__ vt) {
  int i = blockIdx.x * 256 + threadIdx.x;
  if (i < BB * NN) {
    int b = i / NN, n = i % NN;
    v4f r;
#pragma unroll
    for (int c = 0; c < CVV; ++c) r[c] = val[((size_t)b * CVV + c) * NN + n];
    *(v4f*)(vt + (size_t)i * CVV) = r;
  }
}

// ---------------- projection from f32 [b][c][n] source (Q = X Wq^T + bq) ----
__global__ __launch_bounds__(128) void k_proj_f32(const float* __restrict__ X,
                                                  const _Float16* __restrict__ W,
                                                  const float* __restrict__ bias,
                                                  _Float16* __restrict__ out) {
  const int lane = threadIdx.x & 31, wave = threadIdx.x >> 5;
  const int col = lane & 15, half = lane >> 4;
  const int b = blockIdx.x / NBLK_PER_B;
  const int n0 = (blockIdx.x % NBLK_PER_B) * ROWS_PER_BLK + wave * 16;
  const int row = n0 + col;

  v16h a[4];
  const float* xb = X + (size_t)b * CC * NN;
#pragma unroll
  for (int q = 0; q < 4; ++q)
#pragma unroll
    for (int j = 0; j < 16; ++j) {
      int cidx = q * 32 + (j >> 3) * 16 + half * 8 + (j & 7);
      a[q][j] = (_Float16)xb[(size_t)cidx * NN + row];
    }

  _Float16* ob = out + (size_t)b * NN * CC;
#pragma unroll 1
  for (int d0 = 0; d0 < CC; d0 += 16) {
    v8f c = {};
#pragma unroll
    for (int q = 0; q < 4; ++q) {
      v8h blo = *(const v8h*)(W + (size_t)(d0 + col) * CC + q * 32 + half * 8);
      v8h bhi = *(const v8h*)(W + (size_t)(d0 + col) * CC + q * 32 + 16 + half * 8);
      v16h bmat = __builtin_shufflevector(blo, bhi, 0,1,2,3,4,5,6,7,8,9,10,11,12,13,14,15);
      c = __builtin_amdgcn_wmma_f32_16x16x32_f16(false, a[q], false, bmat, (short)0, c, false, false);
    }
    float bv = bias[d0 + col];
#pragma unroll
    for (int r = 0; r < 8; ++r)
      ob[(size_t)(n0 + half * 8 + r) * CC + d0 + col] = (_Float16)(c[r] + bv);
  }
}

// ---------------- projection from f16 row-major source (K = Q Wk^T + bk) ----
__global__ __launch_bounds__(128) void k_proj_f16(const _Float16* __restrict__ X16,
                                                  const _Float16* __restrict__ W,
                                                  const float* __restrict__ bias,
                                                  _Float16* __restrict__ out) {
  const int lane = threadIdx.x & 31, wave = threadIdx.x >> 5;
  const int col = lane & 15, half = lane >> 4;
  const int b = blockIdx.x / NBLK_PER_B;
  const int n0 = (blockIdx.x % NBLK_PER_B) * ROWS_PER_BLK + wave * 16;

  v16h a[4];
  load_a_rowmajor(X16 + ((size_t)b * NN + n0 + col) * CC, half, a);

  _Float16* ob = out + (size_t)b * NN * CC;
#pragma unroll 1
  for (int d0 = 0; d0 < CC; d0 += 16) {
    v8f c = {};
#pragma unroll
    for (int q = 0; q < 4; ++q) {
      v8h blo = *(const v8h*)(W + (size_t)(d0 + col) * CC + q * 32 + half * 8);
      v8h bhi = *(const v8h*)(W + (size_t)(d0 + col) * CC + q * 32 + 16 + half * 8);
      v16h bmat = __builtin_shufflevector(blo, bhi, 0,1,2,3,4,5,6,7,8,9,10,11,12,13,14,15);
      c = __builtin_amdgcn_wmma_f32_16x16x32_f16(false, a[q], false, bmat, (short)0, c, false, false);
    }
    float bv = bias[d0 + col];
#pragma unroll
    for (int r = 0; r < 8; ++r)
      ob[(size_t)(n0 + half * 8 + r) * CC + d0 + col] = (_Float16)(c[r] + bv);
  }
}

// ---------------- fused flash attention: softmax(Q K^T / sqrt(c)) V ---------
__global__ __launch_bounds__(128) void k_attn(const _Float16* __restrict__ Q,
                                              const _Float16* __restrict__ K,
                                              const float* __restrict__ VT,
                                              float* __restrict__ out) {
  __shared__ __align__(64) _Float16 tile[2][GROUP * CC];  // 2 x 16KB, double buffer

  const int lane = threadIdx.x & 31, wave = threadIdx.x >> 5;
  const int col = lane & 15, half = lane >> 4;
  const int b = blockIdx.x / NBLK_PER_B;
  const int m0 = (blockIdx.x % NBLK_PER_B) * ROWS_PER_BLK + wave * 16;

  // resident Q A-operands for this wave's 16 query rows
  v16h a[4];
  load_a_rowmajor(Q + ((size_t)b * NN + m0 + col) * CC, half, a);

  const unsigned long long kbytes =
      (unsigned long long)(K + (size_t)b * NN * CC);
  const unsigned ldsbase = (unsigned)(unsigned long long)(&tile[0][0]);

  // async-stage one 64-key x 128-dim f16 tile (16KB); each wave copies 4KB
  auto stage = [&](int g, int buf) {
    unsigned long long src = kbytes + (unsigned long long)g * (GROUP * CC * 2);
    unsigned off = (unsigned)(wave * 4096 + lane * 16);
    unsigned lds = ldsbase + (unsigned)buf * (GROUP * CC * 2) + off;
#pragma unroll
    for (int i = 0; i < 8; ++i) {
      asm volatile("global_load_async_to_lds_b128 %0, %1, %2"
                   :: "v"(lds), "v"(off), "s"(src) : "memory");
      off += 512u;
      lds += 512u;
    }
  };

  float mrun[8], lrun[8], acc[8][CVV];
#pragma unroll
  for (int r = 0; r < 8; ++r) {
    mrun[r] = -__builtin_inff();
    lrun[r] = 0.f;
#pragma unroll
    for (int c = 0; c < CVV; ++c) acc[r][c] = 0.f;
  }

  stage(0, 0);

  for (int g = 0; g < NGROUPS; ++g) {
    const int cur = g & 1;
    if (g + 1 < NGROUPS) {
      stage(g + 1, cur ^ 1);
      asm volatile("s_wait_asynccnt 0x8" ::: "memory");  // 8 newest still allowed in flight
    } else {
      asm volatile("s_wait_asynccnt 0x0" ::: "memory");
    }
    __syncthreads();

    const _Float16* tl = &tile[cur][0];

    // V rows for this lane's 4 key columns
    v4f vv[4];
#pragma unroll
    for (int t = 0; t < 4; ++t)
      vv[t] = *(const v4f*)(VT + ((size_t)b * NN + g * GROUP + t * 16 + col) * CVV);

    // 4 score tiles of 16x16 (K-dim 128 = 4 chained WMMAs each)
    v8f st[4];
#pragma unroll
    for (int t = 0; t < 4; ++t) {
      v8f c = {};
#pragma unroll
      for (int q = 0; q < 4; ++q) {
        v8h blo = *(const v8h*)(tl + (t * 16 + col) * CC + q * 32 + half * 8);
        v8h bhi = *(const v8h*)(tl + (t * 16 + col) * CC + q * 32 + 16 + half * 8);
        v16h bmat = __builtin_shufflevector(blo, bhi, 0,1,2,3,4,5,6,7,8,9,10,11,12,13,14,15);
        c = __builtin_amdgcn_wmma_f32_16x16x32_f16(false, a[q], false, bmat, (short)0, c, false, false);
      }
      st[t] = c;
    }

    // online softmax update, one pass per row this lane holds
#pragma unroll
    for (int r = 0; r < 8; ++r) {
      float s0 = st[0][r] * SCALE, s1 = st[1][r] * SCALE;
      float s2 = st[2][r] * SCALE, s3 = st[3][r] * SCALE;
      float tm = fmaxf(fmaxf(s0, s1), fmaxf(s2, s3));
      tm = fmaxf(tm, __shfl_xor(tm, 1));
      tm = fmaxf(tm, __shfl_xor(tm, 2));
      tm = fmaxf(tm, __shfl_xor(tm, 4));
      tm = fmaxf(tm, __shfl_xor(tm, 8));
      float mnew = fmaxf(mrun[r], tm);
      float corr = __expf(mrun[r] - mnew);
      mrun[r] = mnew;
      float p0 = __expf(s0 - mnew), p1 = __expf(s1 - mnew);
      float p2 = __expf(s2 - mnew), p3 = __expf(s3 - mnew);
      float ps = p0 + p1 + p2 + p3;
      ps += __shfl_xor(ps, 1);
      ps += __shfl_xor(ps, 2);
      ps += __shfl_xor(ps, 4);
      ps += __shfl_xor(ps, 8);
      lrun[r] = lrun[r] * corr + ps;
#pragma unroll
      for (int c = 0; c < CVV; ++c)
        acc[r][c] = acc[r][c] * corr + p0 * vv[0][c] + p1 * vv[1][c] + p2 * vv[2][c] + p3 * vv[3][c];
    }
    __syncthreads();
  }

  // reduce per-lane partial outputs across the 16 columns, then write [b][cv][n]
#pragma unroll
  for (int r = 0; r < 8; ++r)
#pragma unroll
    for (int c = 0; c < CVV; ++c) {
      float av = acc[r][c];
      av += __shfl_xor(av, 1);
      av += __shfl_xor(av, 2);
      av += __shfl_xor(av, 4);
      av += __shfl_xor(av, 8);
      acc[r][c] = av;
    }
  if (col < CVV) {
#pragma unroll
    for (int r = 0; r < 8; ++r)
      out[((size_t)b * CVV + col) * NN + m0 + half * 8 + r] = acc[r][col] / lrun[r];
  }
}

extern "C" void kernel_launch(void* const* d_in, const int* in_sizes, int n_in,
                              void* d_out, int out_size, void* d_ws, size_t ws_size,
                              hipStream_t stream) {
  const float* feature0 = (const float*)d_in[0];
  const float* value    = (const float*)d_in[1];
  const float* Wq       = (const float*)d_in[2];
  const float* bq       = (const float*)d_in[3];
  const float* Wk       = (const float*)d_in[4];
  const float* bk       = (const float*)d_in[5];
  float* out = (float*)d_out;

  char* ws = (char*)d_ws;
  _Float16* wq16 = (_Float16*)(ws + WQ16_OFF);
  _Float16* wk16 = (_Float16*)(ws + WK16_OFF);
  _Float16* q16  = (_Float16*)(ws + Q16_OFF);
  _Float16* k16  = (_Float16*)(ws + K16_OFF);
  float*    vt   = (float*)(ws + VT_OFF);

  k_cvt_weights<<<64, 256, 0, stream>>>(Wq, Wk, wq16, wk16);
  k_vt<<<(BB * NN + 255) / 256, 256, 0, stream>>>(value, vt);
  k_proj_f32<<<BB * NBLK_PER_B, 128, 0, stream>>>(feature0, wq16, bq, q16);
  k_proj_f16<<<BB * NBLK_PER_B, 128, 0, stream>>>(q16, wk16, bk, k16);
  k_attn<<<BB * NBLK_PER_B, 128, 0, stream>>>(q16, k16, vt, out);
}
